// GenerateProposalsOp_80625126080712
// MI455X (gfx1250) — matrix operand order repaired and checked by reading, not compiled, not run on hardware
//
#include <hip/hip_runtime.h>
#include <cmath>

#define PRE_N     6000
#define POST_N    1000
#define SORT_N    8192
#define KEEP_W    188            // ceil(6000/32)
#define NMS_T     0.7f
#define MIN_SIZE_ 16.0f
#define FEAT_     16.0f
#define BBOX_CLIP 4.135166556742356f   // log(1000/16)

typedef __attribute__((ext_vector_type(16))) _Float16 v16h;
typedef __attribute__((ext_vector_type(8)))  float    v8f;

__device__ __forceinline__ unsigned okey(float f) {
    unsigned u = __float_as_uint(f);
    return (u & 0x80000000u) ? ~u : (u | 0x80000000u);
}

// ---------------- output init: batch index in col 0, zeros elsewhere ----------------
__global__ void k_init_out(float* out, int rows) {
    int r = blockIdx.x * blockDim.x + threadIdx.x;
    if (r < rows) {
        out[r * 6 + 0] = (float)(r / POST_N);
        out[r * 6 + 1] = 0.f; out[r * 6 + 2] = 0.f; out[r * 6 + 3] = 0.f;
        out[r * 6 + 4] = 0.f; out[r * 6 + 5] = 0.f;
    }
}

// ---------------- per-image scratch reset ----------------
__global__ void k_prep(unsigned long long* buf, unsigned* hist, unsigned* state, unsigned* counter) {
    int t = blockIdx.x * blockDim.x + threadIdx.x;
    if (t < SORT_N) buf[t] = 0ULL;
    if (t < 256)    hist[t] = 0u;
    if (t == 0) { state[0] = 0u; state[1] = PRE_N; state[2] = 0u; counter[0] = 0u; }
}

// ---------------- MSB-first radix-select histogram (8-bit digits, 4 passes) ----------------
__global__ void k_hist(const float* __restrict__ sc, int n, int pass,
                       const unsigned* __restrict__ state, unsigned* __restrict__ hist) {
    __shared__ unsigned h[256];
    if (threadIdx.x < 256) h[threadIdx.x] = 0u;
    __syncthreads();
    unsigned long long prefix = (unsigned long long)state[0];
    int shift = 32 - 8 * pass;     // 32,24,16,8 — 64-bit shift, defined at 32
    int dsh   = 24 - 8 * pass;
    int gtid   = blockIdx.x * blockDim.x + threadIdx.x;
    int stride = gridDim.x * blockDim.x;
    if ((((size_t)sc) & 15) == 0) {            // b128 streaming path
        const float4* s4 = (const float4*)sc;
        int n4 = n >> 2;
        for (int i = gtid; i < n4; i += stride) {
            __builtin_prefetch(s4 + i + 1024, 0, 1);   // global_prefetch_b8
            float4 v = s4[i];
            unsigned k0 = okey(v.x), k1 = okey(v.y), k2 = okey(v.z), k3 = okey(v.w);
            if (((unsigned long long)k0 >> shift) == prefix) atomicAdd(&h[(k0 >> dsh) & 255u], 1u);
            if (((unsigned long long)k1 >> shift) == prefix) atomicAdd(&h[(k1 >> dsh) & 255u], 1u);
            if (((unsigned long long)k2 >> shift) == prefix) atomicAdd(&h[(k2 >> dsh) & 255u], 1u);
            if (((unsigned long long)k3 >> shift) == prefix) atomicAdd(&h[(k3 >> dsh) & 255u], 1u);
        }
        int base = n4 << 2;
        if (gtid < n - base) {
            unsigned k = okey(sc[base + gtid]);
            if (((unsigned long long)k >> shift) == prefix) atomicAdd(&h[(k >> dsh) & 255u], 1u);
        }
    } else {
        for (int i = gtid; i < n; i += stride) {
            unsigned k = okey(sc[i]);
            if (((unsigned long long)k >> shift) == prefix) atomicAdd(&h[(k >> dsh) & 255u], 1u);
        }
    }
    __syncthreads();
    if (threadIdx.x < 256 && h[threadIdx.x]) atomicAdd(&hist[threadIdx.x], h[threadIdx.x]);
}

// ---------------- pick digit, refine rank, zero histogram ----------------
__global__ void k_pick(unsigned* state, unsigned* hist, int pass) {
    unsigned rank = state[1];
    unsigned prefix = state[0];
    unsigned cum = 0, d = 0;
    for (int dd = 255; dd >= 0; --dd) {
        unsigned c = hist[dd];
        if (cum + c >= rank) { d = (unsigned)dd; break; }
        cum += c;
    }
    prefix = (prefix << 8) | d;
    state[0] = prefix;
    state[1] = rank - cum;
    if (pass == 3) state[2] = prefix;   // K = exact 6000th-largest key
    for (int i = 0; i < 256; ++i) hist[i] = 0u;
}

// ---------------- compact keys >= K as composite (key<<32 | ~idx) ----------------
__global__ void k_compact(const float* __restrict__ sc, int n,
                          const unsigned* __restrict__ state,
                          unsigned long long* __restrict__ buf, unsigned* counter) {
    unsigned K = state[2];
    int gtid   = blockIdx.x * blockDim.x + threadIdx.x;
    int stride = gridDim.x * blockDim.x;
    if ((((size_t)sc) & 15) == 0) {            // b128 streaming path
        const float4* s4 = (const float4*)sc;
        int n4 = n >> 2;
        for (int i = gtid; i < n4; i += stride) {
            __builtin_prefetch(s4 + i + 1024, 0, 1);
            float4 v = s4[i];
            unsigned kk[4] = { okey(v.x), okey(v.y), okey(v.z), okey(v.w) };
#pragma unroll 4
            for (int c = 0; c < 4; ++c) {
                if (kk[c] >= K) {
                    unsigned pos = atomicAdd(counter, 1u);
                    unsigned idx = (unsigned)(4 * i + c);
                    if (pos < SORT_N)
                        buf[pos] = ((unsigned long long)kk[c] << 32) |
                                   (unsigned long long)(0xFFFFFFFFu - idx);
                }
            }
        }
        int base = n4 << 2;
        if (gtid < n - base) {
            unsigned idx = (unsigned)(base + gtid);
            unsigned k = okey(sc[idx]);
            if (k >= K) {
                unsigned pos = atomicAdd(counter, 1u);
                if (pos < SORT_N)
                    buf[pos] = ((unsigned long long)k << 32) |
                               (unsigned long long)(0xFFFFFFFFu - idx);
            }
        }
    } else {
        for (int i = gtid; i < n; i += stride) {
            unsigned k = okey(sc[i]);
            if (k >= K) {
                unsigned pos = atomicAdd(counter, 1u);
                if (pos < SORT_N)
                    buf[pos] = ((unsigned long long)k << 32) |
                               (unsigned long long)(0xFFFFFFFFu - (unsigned)i);
            }
        }
    }
}

// ---------------- single-workgroup bitonic sort, descending, 8192 u64 keys ----------------
__global__ __launch_bounds__(1024) void k_bitonic(unsigned long long* data) {
    __shared__ unsigned long long s[SORT_N];   // 64 KB LDS (WGP has 320 KB)
    int tid = threadIdx.x;
    for (int i = tid; i < SORT_N; i += 1024) s[i] = data[i];
    __syncthreads();
    for (int k = 2; k <= SORT_N; k <<= 1) {
        for (int j = k >> 1; j > 0; j >>= 1) {
            for (int i = tid; i < SORT_N; i += 1024) {
                int ixj = i ^ j;
                if (ixj > i) {
                    unsigned long long a = s[i], b = s[ixj];
                    bool desc = (i & k) == 0;
                    if (desc ? (a < b) : (a > b)) { s[i] = b; s[ixj] = a; }
                }
            }
            __syncthreads();
        }
    }
    for (int i = tid; i < SORT_N; i += 1024) data[i] = s[i];
}

// ---------------- decode, clip, validity; build second sort key ----------------
__global__ void k_decode(const unsigned long long* __restrict__ buf,
                         const float* __restrict__ pred, const float* __restrict__ iminfo,
                         const float* __restrict__ anchors, int A, int H, int W,
                         unsigned long long* __restrict__ key2,
                         float* __restrict__ boxes, float* __restrict__ scores) {
    int p = blockIdx.x * blockDim.x + threadIdx.x;
    if (p >= SORT_N) return;
    if (p >= PRE_N) { key2[p] = (unsigned long long)(SORT_N - 1 - p); return; }
    unsigned long long comp = buf[p];
    unsigned key = (unsigned)(comp >> 32);
    unsigned idx = 0xFFFFFFFFu - (unsigned)(comp & 0xFFFFFFFFu);
    unsigned sb  = (key & 0x80000000u) ? (key ^ 0x80000000u) : ~key;
    float score  = __uint_as_float(sb);

    int a = idx % A;
    int cell = idx / A;
    int w = cell % W;
    int h = cell / W;
    float sx = w * FEAT_, sy = h * FEAT_;
    float ax1 = anchors[4 * a + 0] + sx;
    float ay1 = anchors[4 * a + 1] + sy;
    float ax2 = anchors[4 * a + 2] + sx;
    float ay2 = anchors[4 * a + 3] + sy;
    float aw = ax2 - ax1 + 1.f, ah = ay2 - ay1 + 1.f;
    float cx = ax1 + 0.5f * aw, cy = ay1 + 0.5f * ah;

    size_t HW = (size_t)H * W;
    const float* dp = pred + (size_t)(4 * a) * HW + (size_t)h * W + w;
    float dx = dp[0];
    float dy = dp[HW];
    float dw = fminf(dp[2 * HW], BBOX_CLIP);
    float dh = fminf(dp[3 * HW], BBOX_CLIP);

    float pcx = dx * aw + cx, pcy = dy * ah + cy;
    float pw = expf(dw) * aw, ph = expf(dh) * ah;
    float x1 = pcx - 0.5f * pw, y1 = pcy - 0.5f * ph;
    float x2 = pcx + 0.5f * pw - 1.f, y2 = pcy + 0.5f * ph - 1.f;

    float imh = iminfo[0], imw = iminfo[1], ims = iminfo[2];
    x1 = fminf(fmaxf(x1, 0.f), imw - 1.f);
    y1 = fminf(fmaxf(y1, 0.f), imh - 1.f);
    x2 = fminf(fmaxf(x2, 0.f), imw - 1.f);
    y2 = fminf(fmaxf(y2, 0.f), imh - 1.f);

    float bw = x2 - x1 + 1.f, bh = y2 - y1 + 1.f;
    float xc = x1 + 0.5f * bw, yc = y1 + 0.5f * bh;
    float msz = MIN_SIZE_ * ims;
    bool valid = (bw >= msz) && (bh >= msz) && (xc < imw) && (yc < imh);

    boxes[p * 4 + 0] = x1; boxes[p * 4 + 1] = y1;
    boxes[p * 4 + 2] = x2; boxes[p * 4 + 3] = y2;
    scores[p] = score;

    unsigned long long hi = valid ? ((unsigned long long)key + 1ULL) : 0ULL;
    key2[p] = (hi << 13) | (unsigned long long)(SORT_N - 1 - p);   // stable reorder key
}

// ---------------- gather into final sorted order ----------------
__global__ void k_gather(const unsigned long long* __restrict__ key2,
                         const float* __restrict__ boxes, const float* __restrict__ scores,
                         float* __restrict__ boxesS, float* __restrict__ scoresS,
                         unsigned* __restrict__ validS) {
    int p = blockIdx.x * blockDim.x + threadIdx.x;
    if (p >= PRE_N) return;
    unsigned long long k = key2[p];
    int pos = (SORT_N - 1) - (int)(k & 0x1FFFULL);
    boxesS[p * 4 + 0] = boxes[pos * 4 + 0];
    boxesS[p * 4 + 1] = boxes[pos * 4 + 1];
    boxesS[p * 4 + 2] = boxes[pos * 4 + 2];
    boxesS[p * 4 + 3] = boxes[pos * 4 + 3];
    scoresS[p] = scores[pos];
    validS[p] = (k >> 13) != 0ULL ? 1u : 0u;
}

// ---------------- sequential greedy NMS, one workgroup, boxes in registers ----------------
__global__ __launch_bounds__(1024) void k_nms(const float* __restrict__ boxesS,
                                              const unsigned* __restrict__ validS,
                                              unsigned* __restrict__ keepOut) {
    __shared__ unsigned keep[KEEP_W];
    __shared__ float sB[5];
    int tid = threadIdx.x;
    for (int w = tid; w < KEEP_W; w += 1024) {
        unsigned m = 0;
        int base = w * 32;
        for (int b = 0; b < 32; ++b) {
            int p = base + b;
            if (p < PRE_N && validS[p]) m |= (1u << b);
        }
        keep[w] = m;
    }
    float x1[6], y1[6], x2[6], y2[6], ar[6];
#pragma unroll
    for (int s = 0; s < 6; ++s) {
        int p = tid + s * 1024;
        if (p < PRE_N) {
            x1[s] = boxesS[p * 4 + 0]; y1[s] = boxesS[p * 4 + 1];
            x2[s] = boxesS[p * 4 + 2]; y2[s] = boxesS[p * 4 + 3];
            ar[s] = (x2[s] - x1[s] + 1.f) * (y2[s] - y1[s] + 1.f);
        } else { x1[s] = 0.f; y1[s] = 0.f; x2[s] = -1.f; y2[s] = -1.f; ar[s] = 0.f; }
    }
    __syncthreads();
    for (int i = 0; i < PRE_N; ++i) {
        if (tid == 0) {
            float a0 = boxesS[i * 4 + 0], a1 = boxesS[i * 4 + 1];
            float a2 = boxesS[i * 4 + 2], a3 = boxesS[i * 4 + 3];
            sB[0] = a0; sB[1] = a1; sB[2] = a2; sB[3] = a3;
            sB[4] = (a2 - a0 + 1.f) * (a3 - a1 + 1.f);
        }
        __syncthreads();
        bool keepI = (keep[i >> 5] >> (i & 31)) & 1u;
        if (keepI) {
            float bx1 = sB[0], by1 = sB[1], bx2 = sB[2], by2 = sB[3], bar = sB[4];
#pragma unroll
            for (int s = 0; s < 6; ++s) {
                int p = tid + s * 1024;
                if (p > i && p < PRE_N) {
                    float iw = fminf(x2[s], bx2) - fmaxf(x1[s], bx1) + 1.f;
                    float ih = fminf(y2[s], by2) - fmaxf(y1[s], by1) + 1.f;
                    iw = fmaxf(iw, 0.f); ih = fmaxf(ih, 0.f);
                    float inter = iw * ih;
                    float iou = inter / (ar[s] + bar - inter);
                    if (iou > NMS_T) atomicAnd(&keep[p >> 5], ~(1u << (p & 31)));
                }
            }
        }
        __syncthreads();
    }
    for (int w = tid; w < KEEP_W; w += 1024) keepOut[w] = keep[w];
}

// ---------------- rank + output: prefix of keep-word popcounts via V_WMMA triangular GEMV ----------------
__global__ __launch_bounds__(1024) void k_output(const unsigned* __restrict__ keepW,
                                                 const float* __restrict__ boxesS,
                                                 const float* __restrict__ scoresS,
                                                 float* __restrict__ out, int img) {
    __shared__ float cntf[192];
    __shared__ float inclRaw[192];
    __shared__ float incl[192];
    int tid = threadIdx.x;
    if (tid < 192) cntf[tid] = (tid < KEEP_W) ? (float)__popc(keepW[tid]) : 0.f;
    __syncthreads();

    if (tid < 32) {                       // one full wave: EXEC all ones at the WMMAs
        int lane = tid;
        int row = lane & 15;
        bool hi = lane >= 16;
        // A = 16x16 lower-triangular ones, zero-padded to 16x32 (contiguous k = 16h + j, B the same)
        v16h Am;
#pragma unroll 16
        for (int j = 0; j < 16; ++j) {
            int K = hi ? j + 16 : j;
            Am[j] = (K < 16 && K <= row) ? (_Float16)1.0f : (_Float16)0.0f;
        }
#pragma unroll
        for (int b = 0; b < 12; ++b) {    // 12 blocks of 16 word-counts (188 padded to 192)
            v16h Bm;
#pragma unroll 16
            for (int j = 0; j < 16; ++j) {
                float v = cntf[b * 16 + j];          // unconditional ds_load
                Bm[j] = (_Float16)(hi ? 0.f : v);    // v_cndmask on the value
            }
            v8f Cm = {};
            // D[m][n] = sum_{k<=m} cnt[16b+k]  (inclusive intra-block prefix; exact in f32)
            Cm = __builtin_amdgcn_wmma_f32_16x16x32_f16(false, Am, false, Bm,
                                                        (short)0, Cm, false, false);
            if ((lane & 15) == 0) {                  // lanes 0 (rows 0..7) and 16 (rows 8..15)
                int base = b * 16 + (hi ? 8 : 0);
#pragma unroll 8
                for (int r = 0; r < 8; ++r) inclRaw[base + r] = Cm[r];
            }
        }
    }
    __syncthreads();
    if (tid == 0) {                       // carry across 16-wide blocks
        float off = 0.f;
        for (int i = 0; i < 192; ++i) {
            float v = inclRaw[i] + off;
            incl[i] = v;
            if ((i & 15) == 15) off = v;
        }
    }
    __syncthreads();

    for (int p = tid; p < PRE_N; p += 1024) {
        int w = p >> 5, bit = p & 31;
        unsigned word = keepW[w];
        if ((word >> bit) & 1u) {
            int rank = (int)(w ? incl[w - 1] : 0.f) + __popc(word & ((1u << bit) - 1u));
            if (rank < POST_N) {
                size_t base = ((size_t)img * POST_N + rank) * 6;
                out[base + 0] = (float)img;
                out[base + 1] = boxesS[p * 4 + 0];
                out[base + 2] = boxesS[p * 4 + 1];
                out[base + 3] = boxesS[p * 4 + 2];
                out[base + 4] = boxesS[p * 4 + 3];
                out[base + 5] = scoresS[p];
            }
        }
    }
}

// ---------------- host launcher ----------------
extern "C" void kernel_launch(void* const* d_in, const int* in_sizes, int n_in,
                              void* d_out, int out_size, void* d_ws, size_t ws_size,
                              hipStream_t stream) {
    const float* cls     = (const float*)d_in[0];
    const float* pred    = (const float*)d_in[1];
    const float* iminfo  = (const float*)d_in[2];
    const float* anchors = (const float*)d_in[3];

    int N = in_sizes[2] / 3;
    int A = in_sizes[3] / 4;
    int perImg = in_sizes[0] / N;        // A*H*W
    int HW = perImg / A;
    int H = 200, W = 200;
    {
        int s = (int)(sqrtf((float)HW) + 0.5f);
        if (s * s == HW) { H = s; W = s; }
    }

    char* ws = (char*)d_ws;
    size_t off = 0;
    auto alloc = [&](size_t bytes) {
        size_t cur = off;
        off += (bytes + 255) & ~((size_t)255);
        return cur;
    };
    unsigned long long* buf   = (unsigned long long*)(ws + alloc(SORT_N * 8));
    unsigned long long* key2  = (unsigned long long*)(ws + alloc(SORT_N * 8));
    float*    boxes   = (float*)(ws + alloc((size_t)PRE_N * 4 * 4));
    float*    scores  = (float*)(ws + alloc((size_t)PRE_N * 4));
    float*    boxesS  = (float*)(ws + alloc((size_t)PRE_N * 4 * 4));
    float*    scoresS = (float*)(ws + alloc((size_t)PRE_N * 4));
    unsigned* validS  = (unsigned*)(ws + alloc((size_t)PRE_N * 4));
    unsigned* keepW   = (unsigned*)(ws + alloc(KEEP_W * 4));
    unsigned* hist    = (unsigned*)(ws + alloc(256 * 4));
    unsigned* state   = (unsigned*)(ws + alloc(4 * 4));
    unsigned* counter = (unsigned*)(ws + alloc(16));

    int rows = out_size / 6;
    k_init_out<<<(rows + 255) / 256, 256, 0, stream>>>((float*)d_out, rows);

    int n = perImg;
    int hb = ((n >> 2) + 255) / 256;     // sized for the float4 path
    if (hb > 1024) hb = 1024;
    if (hb < 1)    hb = 1;

    for (int img = 0; img < N; ++img) {
        const float* sc = cls  + (size_t)img * perImg;
        const float* pr = pred + (size_t)img * 4 * perImg;
        const float* ii = iminfo + (size_t)img * 3;

        k_prep<<<32, 256, 0, stream>>>(buf, hist, state, counter);
        for (int pass = 0; pass < 4; ++pass) {
            k_hist<<<hb, 256, 0, stream>>>(sc, n, pass, state, hist);
            k_pick<<<1, 1, 0, stream>>>(state, hist, pass);
        }
        k_compact<<<hb, 256, 0, stream>>>(sc, n, state, buf, counter);
        k_bitonic<<<1, 1024, 0, stream>>>(buf);
        k_decode<<<SORT_N / 256, 256, 0, stream>>>(buf, pr, ii, anchors, A, H, W,
                                                   key2, boxes, scores);
        k_bitonic<<<1, 1024, 0, stream>>>(key2);
        k_gather<<<(PRE_N + 255) / 256, 256, 0, stream>>>(key2, boxes, scores,
                                                          boxesS, scoresS, validS);
        k_nms<<<1, 1024, 0, stream>>>(boxesS, validS, keepW);
        k_output<<<1, 1024, 0, stream>>>(keepW, boxesS, scoresS, (float*)d_out, img);
    }
    (void)n_in; (void)ws_size;
}
